// GeoGNNBlock_45947560132977
// MI455X (gfx1250) — matrix operand order repaired and checked by reading, not compiled, not run on hardware
//
#include <hip/hip_runtime.h>
#include <hip/hip_bf16.h>

#define N_NODES 4096
#define E_DIM   256
#define E2_DIM  512
#define N_EDGES 131072
#define G_GROUPS 64
#define H_HEADS 4
#define DH_DIM  64

typedef __attribute__((ext_vector_type(16))) _Float16 v16h;
typedef __attribute__((ext_vector_type(8)))  float    v8f;

// ---------------------------------------------------------------------------
// Tiled WMMA GEMM: C = epi(A[M,K] @ B[K,N] + bias + addC)
// 256 threads = 8 waves (4x2); block tile 128x64, K-step 64.
// Each wave computes a 32x32 C tile = 2x2 WMMA fragments (8 WMMAs/stage).
// Tiles staged fp32 in LDS via CDNA5 async global->LDS (ASYNCcnt), converted
// to f16 at fragment build; fp32 accumulate.
// ---------------------------------------------------------------------------
#define GEMM_BM 128
#define GEMM_BN 64
#define GEMM_BK 64
#define GEMM_PITCH 68  // floats; 16B-aligned rows, bank-conflict-free columns

template <int EPI>
__global__ __launch_bounds__(256) void wmma_gemm(
    const float* __restrict__ A, const float* __restrict__ B,
    const float* __restrict__ bias, const float* __restrict__ addC,
    float* __restrict__ C, int M, int Ncols, int K) {
  __shared__ float As[GEMM_BM][GEMM_PITCH];
  __shared__ float Bs[GEMM_BK][GEMM_PITCH];
  const int tid  = threadIdx.x;
  const int lane = tid & 31;
  const int w    = tid >> 5;
  const int m0   = blockIdx.y * GEMM_BM;
  const int n0   = blockIdx.x * GEMM_BN;
  const int wrow = (w >> 1) * 32;  // 0,32,64,96
  const int wcol = (w & 1) * 32;   // 0,32
  const int h    = lane >> 4;
  const int l16  = lane & 15;

  const unsigned asA = (unsigned)(size_t)&As[0][0];
  const unsigned asB = (unsigned)(size_t)&Bs[0][0];

  v8f acc[2][2];
#pragma unroll
  for (int i = 0; i < 2; ++i)
#pragma unroll
    for (int j = 0; j < 2; ++j)
#pragma unroll
      for (int r = 0; r < 8; ++r) acc[i][j][r] = 0.f;

  for (int k0 = 0; k0 < K; k0 += GEMM_BK) {
    // ---- async stage A tile: 128x64 f32, 16B per lane-transfer ----
    for (int i = tid; i < (GEMM_BM * GEMM_BK) / 4; i += 256) {
      int r = i / (GEMM_BK / 4), c4 = i % (GEMM_BK / 4);
      unsigned long long gp =
          (unsigned long long)(size_t)(A + (size_t)(m0 + r) * K + k0 + c4 * 4);
      unsigned lo = asA + (unsigned)(r * GEMM_PITCH + c4 * 4) * 4u;
      asm volatile("global_load_async_to_lds_b128 %0, %1, off"
                   :: "v"(lo), "v"(gp) : "memory");
    }
    // ---- async stage B tile: 64x64 f32 ----
    for (int i = tid; i < (GEMM_BK * GEMM_BN) / 4; i += 256) {
      int r = i / (GEMM_BN / 4), c4 = i % (GEMM_BN / 4);
      unsigned long long gp =
          (unsigned long long)(size_t)(B + (size_t)(k0 + r) * Ncols + n0 + c4 * 4);
      unsigned lo = asB + (unsigned)(r * GEMM_PITCH + c4 * 4) * 4u;
      asm volatile("global_load_async_to_lds_b128 %0, %1, off"
                   :: "v"(lo), "v"(gp) : "memory");
    }
    asm volatile("s_wait_asynccnt 0x0" ::: "memory");
    __syncthreads();

#pragma unroll
    for (int kk = 0; kk < 2; ++kk) {  // two 32-deep K chunks
      v16h a0, a1, b0, b1;
#pragma unroll
      for (int e = 0; e < 16; ++e) {
        int j = e >> 1, s = e & 1;
        int ka = (j < 4) ? (h * 8 + 2 * j + s) : (16 + h * 8 + 2 * (j - 4) + s);
        a0[e] = (_Float16)As[wrow + l16][kk * 32 + ka];
        a1[e] = (_Float16)As[wrow + 16 + l16][kk * 32 + ka];
        int kb = (j < 4) ? (h * 8 + 2 * j + s) : (16 + h * 8 + 2 * (j - 4) + s);
        b0[e] = (_Float16)Bs[kk * 32 + kb][wcol + l16];
        b1[e] = (_Float16)Bs[kk * 32 + kb][wcol + 16 + l16];
      }
      acc[0][0] = __builtin_amdgcn_wmma_f32_16x16x32_f16(false, a0, false, b0, (short)0, acc[0][0], false, false);
      acc[0][1] = __builtin_amdgcn_wmma_f32_16x16x32_f16(false, a0, false, b1, (short)0, acc[0][1], false, false);
      acc[1][0] = __builtin_amdgcn_wmma_f32_16x16x32_f16(false, a1, false, b0, (short)0, acc[1][0], false, false);
      acc[1][1] = __builtin_amdgcn_wmma_f32_16x16x32_f16(false, a1, false, b1, (short)0, acc[1][1], false, false);
    }
    __syncthreads();
  }

#pragma unroll
  for (int i = 0; i < 2; ++i)
#pragma unroll
    for (int j = 0; j < 2; ++j)
#pragma unroll
      for (int r = 0; r < 8; ++r) {
        int row = m0 + wrow + i * 16 + 8 * h + r;
        int col = n0 + wcol + j * 16 + l16;
        float v = acc[i][j][r];
        if (bias) v += bias[col];
        if (addC) v += addC[(size_t)row * Ncols + col];
        if (EPI == 1) v = fmaxf(v, 0.f);
        C[(size_t)row * Ncols + col] = v;
      }
}

// ---------------------------------------------------------------------------
// Flash attention: one wave per (head, 16-query tile), key tiles of 32.
// ---------------------------------------------------------------------------
__global__ __launch_bounds__(256) void attn_kernel(
    const float* __restrict__ Q, const float* __restrict__ K,
    const float* __restrict__ V, float* __restrict__ O) {
  __shared__ _Float16 Pl[8][16][32];
  const int lane = threadIdx.x & 31;
  const int w    = threadIdx.x >> 5;
  const int tile = blockIdx.x * 8 + w;   // 0 .. H*(N/16)-1
  const int head = tile >> 8;            // tile / (4096/16)
  const int qt   = tile & 255;
  const int h    = lane >> 4;
  const int m    = lane & 15;

  // Q fragments (pre-scaled by 1/sqrt(DH))
  v16h aq0, aq1;
  {
    const float* qp = Q + (size_t)(qt * 16 + m) * E_DIM + head * DH_DIM;
#pragma unroll
    for (int e = 0; e < 16; ++e) {
      int j = e >> 1, s = e & 1;
      int k = (j < 4) ? (h * 8 + 2 * j + s) : (16 + h * 8 + 2 * (j - 4) + s);
      aq0[e] = (_Float16)(qp[k] * 0.125f);
      aq1[e] = (_Float16)(qp[32 + k] * 0.125f);
    }
  }

  float mrun[8], lrun[8];
  v8f oacc[4];
#pragma unroll
  for (int r = 0; r < 8; ++r) { mrun[r] = -1e30f; lrun[r] = 0.f; }
#pragma unroll
  for (int c = 0; c < 4; ++c)
#pragma unroll
    for (int r = 0; r < 8; ++r) oacc[c][r] = 0.f;

  for (int kb = 0; kb < N_NODES; kb += 32) {
    // ---- S = Q @ K^T for 32 keys (two 16-col tiles) ----
    v8f s0 = {0.f,0.f,0.f,0.f,0.f,0.f,0.f,0.f};
    v8f s1 = {0.f,0.f,0.f,0.f,0.f,0.f,0.f,0.f};
    {
      v16h b0, b1;
      const float* kp0 = K + (size_t)(kb + m) * E_DIM + head * DH_DIM;
      const float* kp1 = K + (size_t)(kb + 16 + m) * E_DIM + head * DH_DIM;
#pragma unroll
      for (int e = 0; e < 16; ++e) {
        int j = e >> 1, s = e & 1;
        int kk = (j < 4) ? (h * 8 + 2 * j + s) : (16 + h * 8 + 2 * (j - 4) + s);
        b0[e] = (_Float16)kp0[kk];
        b1[e] = (_Float16)kp0[32 + kk];
      }
      s0 = __builtin_amdgcn_wmma_f32_16x16x32_f16(false, aq0, false, b0, (short)0, s0, false, false);
      s0 = __builtin_amdgcn_wmma_f32_16x16x32_f16(false, aq1, false, b1, (short)0, s0, false, false);
#pragma unroll
      for (int e = 0; e < 16; ++e) {
        int j = e >> 1, s = e & 1;
        int kk = (j < 4) ? (h * 8 + 2 * j + s) : (16 + h * 8 + 2 * (j - 4) + s);
        b0[e] = (_Float16)kp1[kk];
        b1[e] = (_Float16)kp1[32 + kk];
      }
      s1 = __builtin_amdgcn_wmma_f32_16x16x32_f16(false, aq0, false, b0, (short)0, s1, false, false);
      s1 = __builtin_amdgcn_wmma_f32_16x16x32_f16(false, aq1, false, b1, (short)0, s1, false, false);
    }

    // ---- online softmax; slot r of lane-half h holds row 8h+r ----
    float corr[8];
#pragma unroll
    for (int r = 0; r < 8; ++r) {
      float v = fmaxf(s0[r], s1[r]);
      v = fmaxf(v, __shfl_xor(v, 1, 32));
      v = fmaxf(v, __shfl_xor(v, 2, 32));
      v = fmaxf(v, __shfl_xor(v, 4, 32));
      v = fmaxf(v, __shfl_xor(v, 8, 32));
      float mn = fmaxf(mrun[r], v);
      corr[r] = __expf(mrun[r] - mn);
      mrun[r] = mn;
    }
    float p0[8], p1[8];
#pragma unroll
    for (int r = 0; r < 8; ++r) {
      p0[r] = __expf(s0[r] - mrun[r]);
      p1[r] = __expf(s1[r] - mrun[r]);
      float rs = p0[r] + p1[r];
      rs += __shfl_xor(rs, 1, 32);
      rs += __shfl_xor(rs, 2, 32);
      rs += __shfl_xor(rs, 4, 32);
      rs += __shfl_xor(rs, 8, 32);
      lrun[r] = lrun[r] * corr[r] + rs;
    }
#pragma unroll
    for (int c = 0; c < 4; ++c)
#pragma unroll
      for (int r = 0; r < 8; ++r) oacc[c][r] *= corr[r];

    // ---- restage P (C layout) into A-fragment layout via LDS ----
#pragma unroll
    for (int r = 0; r < 8; ++r) {
      Pl[w][8 * h + r][m]      = (_Float16)p0[r];
      Pl[w][8 * h + r][16 + m] = (_Float16)p1[r];
    }
    __syncthreads();
    v16h ap;
#pragma unroll
    for (int e = 0; e < 16; ++e) {
      int j = e >> 1, s = e & 1;
      int k = (j < 4) ? (h * 8 + 2 * j + s) : (16 + h * 8 + 2 * (j - 4) + s);
      ap[e] = Pl[w][m][k];
    }

    // ---- O += P @ V ----
#pragma unroll
    for (int c = 0; c < 4; ++c) {
      v16h bv;
#pragma unroll
      for (int e = 0; e < 16; ++e) {
        int j = e >> 1, s = e & 1;
        int kk = (j < 4) ? (h * 8 + 2 * j + s) : (16 + h * 8 + 2 * (j - 4) + s);  // key offset within tile
        bv[e] = (_Float16)V[(size_t)(kb + kk) * E_DIM + head * DH_DIM + c * 16 + m];
      }
      oacc[c] = __builtin_amdgcn_wmma_f32_16x16x32_f16(false, ap, false, bv,
                                                       (short)0, oacc[c], false, false);
    }
    __syncthreads();
  }

#pragma unroll
  for (int c = 0; c < 4; ++c)
#pragma unroll
    for (int r = 0; r < 8; ++r) {
      O[(size_t)(qt * 16 + 8 * h + r) * E_DIM + head * DH_DIM + c * 16 + m] =
          oacc[c][r] / lrun[r];
    }
}

// ---------------------------------------------------------------------------
// Edge / elementwise / norm kernels
// ---------------------------------------------------------------------------
__global__ __launch_bounds__(256) void gine_edge(
    const float* __restrict__ node, const float* __restrict__ eh,
    const int* __restrict__ ei, float* __restrict__ agg) {
  int e = blockIdx.x, c = threadIdx.x;
  int s = ei[e], d = ei[N_EDGES + e];
  float msg = fmaxf(node[(size_t)s * E_DIM + c] + eh[(size_t)e * E_DIM + c], 0.f);
  atomicAdd(&agg[(size_t)d * E_DIM + c], msg);
}

__global__ void add_inplace(float* __restrict__ acc, const float* __restrict__ x, int n) {
  int i = blockIdx.x * blockDim.x + threadIdx.x;
  if (i < n) acc[i] += x[i];
}

__global__ __launch_bounds__(256) void dir_edge(
    const float* __restrict__ pos, const float* __restrict__ gnn,
    const int* __restrict__ ei, const float* __restrict__ dw,
    const float* __restrict__ db, float* __restrict__ sagg) {
  int e = blockIdx.x, c = threadIdx.x;
  int s = ei[e], d = ei[N_EDGES + e];
  float vx = pos[d * 3 + 0] - pos[s * 3 + 0];
  float vy = pos[d * 3 + 1] - pos[s * 3 + 1];
  float vz = pos[d * 3 + 2] - pos[s * 3 + 2];
  float inv = 1.f / (sqrtf(vx * vx + vy * vy + vz * vz) + 1e-8f);
  vx *= inv; vy *= inv; vz *= inv;
  float df = vx * dw[c] + vy * dw[E_DIM + c] + vz * dw[2 * E_DIM + c] + db[c];
  atomicAdd(&sagg[(size_t)d * E_DIM + c], gnn[(size_t)s * E_DIM + c] * df);
}

__global__ __launch_bounds__(256) void ln_kernel(
    const float* __restrict__ X, const float* __restrict__ g,
    const float* __restrict__ b, float* __restrict__ Y, int Ncols, int relu) {
  __shared__ float red[256];
  int row = blockIdx.x;
  const float* x = X + (size_t)row * Ncols;
  float s = 0.f, s2 = 0.f;
  for (int c = threadIdx.x; c < Ncols; c += 256) { float v = x[c]; s += v; s2 += v * v; }
  red[threadIdx.x] = s; __syncthreads();
  for (int o = 128; o > 0; o >>= 1) { if (threadIdx.x < o) red[threadIdx.x] += red[threadIdx.x + o]; __syncthreads(); }
  float mean = red[0] / Ncols; __syncthreads();
  red[threadIdx.x] = s2; __syncthreads();
  for (int o = 128; o > 0; o >>= 1) { if (threadIdx.x < o) red[threadIdx.x] += red[threadIdx.x + o]; __syncthreads(); }
  float var = red[0] / Ncols - mean * mean;
  float inv = rsqrtf(var + 1e-5f);
  for (int c = threadIdx.x; c < Ncols; c += 256) {
    float v = g[c] * (x[c] - mean) * inv + b[c];
    if (relu) v = fmaxf(v, 0.f);
    Y[(size_t)row * Ncols + c] = v;
  }
}

__global__ __launch_bounds__(256) void blend_ln(
    const float* __restrict__ gl, const float* __restrict__ fus,
    const float* __restrict__ idn, const float* __restrict__ g,
    const float* __restrict__ b, float* __restrict__ Y) {
  __shared__ float red[256];
  int row = blockIdx.x, c = threadIdx.x;
  size_t idx = (size_t)row * E_DIM + c;
  float gate = 1.f / (1.f + __expf(-gl[idx]));
  float t = gate * fus[idx] + (1.f - gate) * idn[idx];
  red[c] = t; __syncthreads();
  for (int o = 128; o > 0; o >>= 1) { if (c < o) red[c] += red[c + o]; __syncthreads(); }
  float mean = red[0] / E_DIM; __syncthreads();
  red[c] = t * t; __syncthreads();
  for (int o = 128; o > 0; o >>= 1) { if (c < o) red[c] += red[c + o]; __syncthreads(); }
  float var = red[0] / E_DIM - mean * mean;
  Y[idx] = g[c] * (t - mean) * rsqrtf(var + 1e-5f) + b[c];
}

__global__ void gn_count(const int* __restrict__ nid, float* __restrict__ cnt) {
  int n = blockIdx.x * blockDim.x + threadIdx.x;
  if (n < N_NODES) atomicAdd(&cnt[nid[n]], 1.f);
}

__global__ __launch_bounds__(256) void gn_sum(
    const float* __restrict__ y, const int* __restrict__ nid, float* __restrict__ gsum) {
  int n = blockIdx.x, c = threadIdx.x;
  atomicAdd(&gsum[(size_t)nid[n] * E_DIM + c], y[(size_t)n * E_DIM + c]);
}

__global__ void gn_div(float* __restrict__ buf, const float* __restrict__ cnt, int n) {
  int i = blockIdx.x * blockDim.x + threadIdx.x;
  if (i < n) buf[i] = buf[i] / fmaxf(cnt[i / E_DIM], 1.f);
}

__global__ __launch_bounds__(256) void gn_var(
    const float* __restrict__ y, const int* __restrict__ nid,
    const float* __restrict__ mean, float* __restrict__ gvar) {
  int n = blockIdx.x, c = threadIdx.x;
  int g = nid[n];
  float cen = y[(size_t)n * E_DIM + c] - mean[(size_t)g * E_DIM + c];
  atomicAdd(&gvar[(size_t)g * E_DIM + c], cen * cen);
}

__global__ __launch_bounds__(256) void gn_final(
    const float* __restrict__ y, const int* __restrict__ nid,
    const float* __restrict__ mean, const float* __restrict__ var,
    const float* __restrict__ gg, const float* __restrict__ gb,
    float* __restrict__ out) {
  int n = blockIdx.x, c = threadIdx.x;
  int g = nid[n];
  float cen = y[(size_t)n * E_DIM + c] - mean[(size_t)g * E_DIM + c];
  float v = gg[c] * cen * rsqrtf(var[(size_t)g * E_DIM + c] + 1e-5f) + gb[c];
  out[(size_t)n * E_DIM + c] = fmaxf(v, 0.f);
}

// ---------------------------------------------------------------------------
extern "C" void kernel_launch(void* const* d_in, const int* in_sizes, int n_in,
                              void* d_out, int out_size, void* d_ws, size_t ws_size,
                              hipStream_t stream) {
  const float* node_hidden = (const float*)d_in[0];
  const float* edge_hidden = (const float*)d_in[1];
  const float* pos         = (const float*)d_in[2];
  const float* gine_w1     = (const float*)d_in[3];
  const float* gine_b1     = (const float*)d_in[4];
  const float* gine_ln_g   = (const float*)d_in[5];
  const float* gine_ln_b   = (const float*)d_in[6];
  const float* gine_w2     = (const float*)d_in[7];
  const float* gine_b2     = (const float*)d_in[8];
  const float* norm1_g     = (const float*)d_in[9];
  const float* norm1_b     = (const float*)d_in[10];
  const float* dir_w       = (const float*)d_in[11];
  const float* dir_b       = (const float*)d_in[12];
  const float* msg_w       = (const float*)d_in[13];
  const float* msg_b       = (const float*)d_in[14];
  const float* wq          = (const float*)d_in[15];
  const float* bq          = (const float*)d_in[16];
  const float* wk          = (const float*)d_in[17];
  const float* bk          = (const float*)d_in[18];
  const float* wv          = (const float*)d_in[19];
  const float* bv          = (const float*)d_in[20];
  const float* wo          = (const float*)d_in[21];
  const float* bo          = (const float*)d_in[22];
  const float* gate_w      = (const float*)d_in[23];
  const float* gate_b      = (const float*)d_in[24];
  const float* norm2_g     = (const float*)d_in[25];
  const float* norm2_b     = (const float*)d_in[26];
  const float* gn_g        = (const float*)d_in[27];
  const float* gn_b        = (const float*)d_in[28];
  const int*   edge_index  = (const int*)d_in[29];
  const int*   node_id     = (const int*)d_in[30];

  const size_t NE = (size_t)N_NODES * E_DIM;
  float* ws      = (float*)d_ws;
  float* agg     = ws;                 // [N,E] (becomes h in-place)
  float* t1      = agg + NE;           // [N,2E]
  float* gnn     = t1 + (size_t)N_NODES * E2_DIM;
  float* sagg    = gnn + NE;
  float* spatial = sagg + NE;
  float* qb      = spatial + NE;
  float* kbuf    = qb + NE;
  float* vbuf    = kbuf + NE;
  float* attn    = vbuf + NE;
  float* fusedb  = attn + NE;
  float* gatelin = fusedb + NE;
  float* yb      = gatelin + NE;
  float* gsum    = yb + NE;            // [G,E]
  float* gvar    = gsum + (size_t)G_GROUPS * E_DIM;
  float* gcnt    = gvar + (size_t)G_GROUPS * E_DIM;  // [G]

  hipMemsetAsync(agg, 0, NE * sizeof(float), stream);
  hipMemsetAsync(sagg, 0, NE * sizeof(float), stream);
  hipMemsetAsync(gsum, 0, (2 * (size_t)G_GROUPS * E_DIM + G_GROUPS) * sizeof(float), stream);

  // --- GINEConv ---
  gine_edge<<<N_EDGES, 256, 0, stream>>>(node_hidden, edge_hidden, edge_index, agg);
  add_inplace<<<(int)((NE + 255) / 256), 256, 0, stream>>>(agg, node_hidden, (int)NE);
  wmma_gemm<0><<<dim3(E2_DIM / GEMM_BN, N_NODES / GEMM_BM), 256, 0, stream>>>(
      agg, gine_w1, gine_b1, nullptr, t1, N_NODES, E2_DIM, E_DIM);
  ln_kernel<<<N_NODES, 256, 0, stream>>>(t1, gine_ln_g, gine_ln_b, t1, E2_DIM, 1);
  wmma_gemm<0><<<dim3(E_DIM / GEMM_BN, N_NODES / GEMM_BM), 256, 0, stream>>>(
      t1, gine_w2, gine_b2, nullptr, gnn, N_NODES, E_DIM, E2_DIM);
  ln_kernel<<<N_NODES, 256, 0, stream>>>(gnn, norm1_g, norm1_b, gnn, E_DIM, 0);

  // --- Directional message passing ---
  dir_edge<<<N_EDGES, 256, 0, stream>>>(pos, gnn, edge_index, dir_w, dir_b, sagg);
  wmma_gemm<0><<<dim3(E_DIM / GEMM_BN, N_NODES / GEMM_BM), 256, 0, stream>>>(
      sagg, msg_w, msg_b, nullptr, spatial, N_NODES, E_DIM, E_DIM);

  // --- MHA ---
  wmma_gemm<0><<<dim3(E_DIM / GEMM_BN, N_NODES / GEMM_BM), 256, 0, stream>>>(
      gnn, wq, bq, nullptr, qb, N_NODES, E_DIM, E_DIM);
  wmma_gemm<0><<<dim3(E_DIM / GEMM_BN, N_NODES / GEMM_BM), 256, 0, stream>>>(
      spatial, wk, bk, nullptr, kbuf, N_NODES, E_DIM, E_DIM);
  wmma_gemm<0><<<dim3(E_DIM / GEMM_BN, N_NODES / GEMM_BM), 256, 0, stream>>>(
      spatial, wv, bv, nullptr, vbuf, N_NODES, E_DIM, E_DIM);
  attn_kernel<<<(H_HEADS * (N_NODES / 16)) / 8, 256, 0, stream>>>(qb, kbuf, vbuf, attn);
  wmma_gemm<0><<<dim3(E_DIM / GEMM_BN, N_NODES / GEMM_BM), 256, 0, stream>>>(
      attn, wo, bo, nullptr, fusedb, N_NODES, E_DIM, E_DIM);

  // --- gated residual: concat GEMM split into two GEMMs ---
  wmma_gemm<0><<<dim3(E_DIM / GEMM_BN, N_NODES / GEMM_BM), 256, 0, stream>>>(
      fusedb, gate_w, gate_b, nullptr, gatelin, N_NODES, E_DIM, E_DIM);
  wmma_gemm<0><<<dim3(E_DIM / GEMM_BN, N_NODES / GEMM_BM), 256, 0, stream>>>(
      node_hidden, gate_w + (size_t)E_DIM * E_DIM, nullptr, gatelin, gatelin,
      N_NODES, E_DIM, E_DIM);
  blend_ln<<<N_NODES, 256, 0, stream>>>(gatelin, fusedb, node_hidden, norm2_g, norm2_b, yb);

  // --- GraphNorm ---
  gn_count<<<(N_NODES + 255) / 256, 256, 0, stream>>>(node_id, gcnt);
  gn_sum<<<N_NODES, 256, 0, stream>>>(yb, node_id, gsum);
  gn_div<<<(G_GROUPS * E_DIM + 255) / 256, 256, 0, stream>>>(gsum, gcnt, G_GROUPS * E_DIM);
  gn_var<<<N_NODES, 256, 0, stream>>>(yb, node_id, gsum, gvar);
  gn_div<<<(G_GROUPS * E_DIM + 255) / 256, 256, 0, stream>>>(gvar, gcnt, G_GROUPS * E_DIM);
  gn_final<<<N_NODES, 256, 0, stream>>>(yb, node_id, gsum, gvar, gn_g, gn_b, (float*)d_out);
}
